// DistanceLoss_31860067401818
// MI455X (gfx1250) — matrix operand order repaired and checked by reading, not compiled, not run on hardware
//
#include <hip/hip_runtime.h>
#include <math.h>

// ---------------------------------------------------------------------------
// DistanceLoss pipeline for MI455X (gfx1250, wave32, WMMA).
// Heavy GEMMs: bf16x3-split (hi/lo) v_wmma_f32_16x16x32_bf16, f32 accumulate
// (~fp32 accuracy at bf16 matrix-pipe rate).
// This revision: BK=64 (24 WMMA per barrier pair), hoisted gather addressing
// for the embedding GEMM, batched global loads before convert/store.
// ---------------------------------------------------------------------------

#define SEQ_LEN 10
#define WAY 5
#define SHOT 5
#define T_TUP 45          // C(10,2)
#define ST 225            // SHOT * T_TUP
#define NQ 75
#define NS 25
#define NQT 3375          // NQ * T_TUP
#define NST 1125          // NS * T_TUP
#define NROWS 4500        // NQT + NST
#define IN_DIM 2048
#define OUT_DIM 2048
#define K_EMBED 4096      // 2 * IN_DIM

#define BM 128            // block tile rows
#define BN 64             // block tile cols
#define BK 64             // k-step (two 32-K WMMA substeps per barrier pair)
#define LDST 72           // LDS row stride in bf16 (144 B: 16B-aligned, conflict-free)

typedef __bf16 bf16;
typedef __attribute__((ext_vector_type(16))) __bf16 v16bf;
typedef __attribute__((ext_vector_type(8)))  __bf16 v8bf;
typedef __attribute__((ext_vector_type(8)))  float  v8f;

// frame-pair table: combinations(range(10), 2) lexicographic
__constant__ int d_pair0[T_TUP] = {0,0,0,0,0,0,0,0,0, 1,1,1,1,1,1,1,1, 2,2,2,2,2,2,2,
                                   3,3,3,3,3,3, 4,4,4,4,4, 5,5,5,5, 6,6,6, 7,7, 8};
__constant__ int d_pair1[T_TUP] = {1,2,3,4,5,6,7,8,9, 2,3,4,5,6,7,8,9, 3,4,5,6,7,8,9,
                                   4,5,6,7,8,9, 5,6,7,8,9, 6,7,8,9, 7,8,9, 8,9, 9};

// ---------------- bf16 split helpers (manual, RNE) -------------------------
__device__ inline unsigned short bfbits(float x) {
  union { float f; unsigned u; } v; v.f = x;
  unsigned u = v.u + 0x7fffu + ((v.u >> 16) & 1u);
  return (unsigned short)(u >> 16);
}
__device__ inline float fb(unsigned short s) {
  union { unsigned u; float f; } o; o.u = ((unsigned)s) << 16;
  return o.f;
}

__device__ inline v8f wmma_bf16(v16bf a, v16bf b, v8f c) {
  return __builtin_amdgcn_wmma_f32_16x16x32_bf16(false, a, false, b, (short)0, c,
                                                 false, false);
}

// ---------------- fragment loads from LDS tiles (16B-aligned) --------------
// A-fragment (16x32 bf16, M x K), K window at bf16 offset ks. ISA 7.12.2:
// lane<16: M=lane, K in {0..7}U{16..23}; lane>=16: M=lane-16, K in {8..15}U{24..31}
__device__ inline v16bf frag_a(const bf16* tile, int r0, int ks) {
  int lane = threadIdx.x & 31;
  int half = lane >> 4;
  const bf16* p = tile + ((lane & 15) + r0) * LDST + ks + half * 8;
  v8bf c0 = *(const v8bf*)p;          // K = half*8 .. +8        (16B aligned)
  v8bf c1 = *(const v8bf*)(p + 16);   // K = 16 + half*8 .. +8   (16B aligned)
  return __builtin_shufflevector(c0, c1, 0, 1, 2, 3, 4, 5, 6, 7,
                                 8, 9, 10, 11, 12, 13, 14, 15);
}
// B-fragment (32x16 bf16, K x N), tile stored as N x K (rows of W / E):
// lane = khalf*16 + n ; K in {khalf*8..+8} U {16+khalf*8..+8} (matches frag_a)
__device__ inline v16bf frag_b(const bf16* tile, int n0, int ks) {
  int lane = threadIdx.x & 31;
  int khalf = lane >> 4;
  const bf16* p = tile + ((lane & 15) + n0) * LDST + ks + khalf * 8;
  v8bf c0 = *(const v8bf*)p;          // 16B aligned
  v8bf c1 = *(const v8bf*)(p + 16);
  return __builtin_shufflevector(c0, c1, 0, 1, 2, 3, 4, 5, 6, 7,
                                 8, 9, 10, 11, 12, 13, 14, 15);
}

// ---------------- global -> LDS tile conversion (fp32 -> hi/lo bf16) -------
__device__ inline void store4(bf16* basep, int idx, unsigned short a,
                              unsigned short b, unsigned short c, unsigned short d) {
  uint2 u;
  u.x = (unsigned)a | ((unsigned)b << 16);
  u.y = (unsigned)c | ((unsigned)d << 16);
  *(uint2*)(basep + idx) = u;   // idx multiple of 4 -> 8B aligned -> ds_store_b64
}

__device__ inline void split_store4(float4 v, bf16* hi, bf16* lo, int idx) {
  unsigned short h0 = bfbits(v.x), h1 = bfbits(v.y), h2 = bfbits(v.z), h3 = bfbits(v.w);
  unsigned short l0 = bfbits(v.x - fb(h0)), l1 = bfbits(v.y - fb(h1));
  unsigned short l2 = bfbits(v.z - fb(h2)), l3 = bfbits(v.w - fb(h3));
  store4(hi, idx, h0, h1, h2, h3);
  store4(lo, idx, l0, l1, l2, l3);
}

// BR x 64 fp32 tile: gather all float4s first, then convert + LDS-store.
// Each iter of 256 threads covers 16 rows (16 float4 per 64-col row).
template <int BR>
__device__ inline void load_tile_plain(const float* src, int ldk, int r0, int k0,
                                       int rmax, bf16* hi, bf16* lo) {
  constexpr int NIT = BR / 16;
  float4 vals[NIT];
  int tid = threadIdx.x;
#pragma unroll
  for (int it = 0; it < NIT; ++it) {
    int e = tid + (it << 8);
    int row = e >> 4;
    int c4 = (e & 15) << 2;
    int gr = r0 + row;
    vals[it] = make_float4(0.f, 0.f, 0.f, 0.f);
    if (gr < rmax) vals[it] = *(const float4*)(src + (size_t)gr * ldk + k0 + c4);
  }
#pragma unroll
  for (int it = 0; it < NIT; ++it) {
    int e = tid + (it << 8);
    int row = e >> 4;
    int c4 = (e & 15) << 2;
    split_store4(vals[it], hi, lo, row * LDST + c4);
  }
}

// Embedding-GEMM A tile: per-thread row base pointers precomputed per K-half
// (frame select + row decomposition hoisted out of the k-loop).
#define NRP (BM / 16)
__device__ inline void embed_row_ptrs(const float* queries, const float* support,
                                      int m0, int halfsel, const float* rp[NRP]) {
  int tid = threadIdx.x;
#pragma unroll
  for (int it = 0; it < NRP; ++it) {
    int e = tid + (it << 8);
    int row = e >> 4;
    int c4 = (e & 15) << 2;
    int gr = m0 + row;
    rp[it] = nullptr;
    if (gr < NROWS) {
      const float* bse;
      int n, t;
      if (gr < NQT) { n = gr / T_TUP; t = gr - n * T_TUP; bse = queries; }
      else { int rr = gr - NQT; n = rr / T_TUP; t = rr - n * T_TUP; bse = support; }
      int frame = halfsel ? d_pair1[t] : d_pair0[t];
      rp[it] = bse + ((size_t)n * SEQ_LEN + frame) * IN_DIM + c4;
    }
  }
}

__device__ inline void load_tile_embed(const float* const rp[NRP], int koff,
                                       bf16* hi, bf16* lo) {
  float4 vals[NRP];
  int tid = threadIdx.x;
#pragma unroll
  for (int it = 0; it < NRP; ++it) {
    vals[it] = make_float4(0.f, 0.f, 0.f, 0.f);
    if (rp[it]) vals[it] = *(const float4*)(rp[it] + koff);
  }
#pragma unroll
  for (int it = 0; it < NRP; ++it) {
    int e = tid + (it << 8);
    int row = e >> 4;
    int c4 = (e & 15) << 2;
    split_store4(vals[it], hi, lo, row * LDST + c4);
  }
}

// ---------------- shared WMMA substep (32x32 per wave, bf16x3) -------------
#define GEMM_SHARED_DECL                                                        \
  __shared__ __align__(16) bf16 sAh[BM * LDST], sAl[BM * LDST];                 \
  __shared__ __align__(16) bf16 sBh[BN * LDST], sBl[BN * LDST];

#define GEMM_WMMA_STEP(wm, wn, ks)                                             \
  {                                                                             \
    v16bf ah0 = frag_a(sAh, (wm) * 32, (ks));                                   \
    v16bf al0 = frag_a(sAl, (wm) * 32, (ks));                                   \
    v16bf ah1 = frag_a(sAh, (wm) * 32 + 16, (ks));                              \
    v16bf al1 = frag_a(sAl, (wm) * 32 + 16, (ks));                              \
    v16bf bh0 = frag_b(sBh, (wn) * 32, (ks));                                   \
    v16bf bl0 = frag_b(sBl, (wn) * 32, (ks));                                   \
    v16bf bh1 = frag_b(sBh, (wn) * 32 + 16, (ks));                              \
    v16bf bl1 = frag_b(sBl, (wn) * 32 + 16, (ks));                              \
    acc00 = wmma_bf16(ah0, bh0, acc00);                                         \
    acc00 = wmma_bf16(ah0, bl0, acc00);                                         \
    acc00 = wmma_bf16(al0, bh0, acc00);                                         \
    acc01 = wmma_bf16(ah0, bh1, acc01);                                         \
    acc01 = wmma_bf16(ah0, bl1, acc01);                                         \
    acc01 = wmma_bf16(al0, bh1, acc01);                                         \
    acc10 = wmma_bf16(ah1, bh0, acc10);                                         \
    acc10 = wmma_bf16(ah1, bl0, acc10);                                         \
    acc10 = wmma_bf16(al1, bh0, acc10);                                         \
    acc11 = wmma_bf16(ah1, bh1, acc11);                                         \
    acc11 = wmma_bf16(ah1, bl1, acc11);                                         \
    acc11 = wmma_bf16(al1, bh1, acc11);                                         \
  }

// ---------------- GEMM 1: embeddings E = relu(gather(X) @ W^T + b) ---------
__global__ void __launch_bounds__(256)
embed_gemm(const float* __restrict__ queries, const float* __restrict__ support,
           const float* __restrict__ W, const float* __restrict__ bias,
           float* __restrict__ E) {
  GEMM_SHARED_DECL
  int m0 = blockIdx.x * BM;
  int n0 = blockIdx.y * BN;
  int wave = threadIdx.x >> 5;
  int wm = wave >> 1;        // 0..3 -> 32-row slab
  int wn = wave & 1;         // 0..1 -> 32-col slab
  v8f acc00 = {0, 0, 0, 0, 0, 0, 0, 0};
  v8f acc01 = acc00, acc10 = acc00, acc11 = acc00;
  const float* rp[NRP];
  for (int h = 0; h < 2; ++h) {            // frame 0 then frame 1 K-half
    embed_row_ptrs(queries, support, m0, h, rp);
    for (int kh = 0; kh < IN_DIM; kh += BK) {
      load_tile_embed(rp, kh, sAh, sAl);
      load_tile_plain<BN>(W, K_EMBED, n0, h * IN_DIM + kh, OUT_DIM, sBh, sBl);
      __syncthreads();
      GEMM_WMMA_STEP(wm, wn, 0)
      GEMM_WMMA_STEP(wm, wn, 32)
      __syncthreads();
    }
  }
  int lane = threadIdx.x & 31;
  int half = lane >> 4, nn = lane & 15;
  int gn0 = n0 + wn * 32 + nn;
  int gn1 = gn0 + 16;
  float b0 = bias[gn0], b1 = bias[gn1];
#pragma unroll
  for (int ms = 0; ms < 2; ++ms) {
    int gm_base = m0 + wm * 32 + ms * 16 + half * 8;
    v8f a0 = ms ? acc10 : acc00;
    v8f a1 = ms ? acc11 : acc01;
#pragma unroll
    for (int v = 0; v < 8; ++v) {
      int gm = gm_base + v;
      if (gm < NROWS) {
        E[(size_t)gm * OUT_DIM + gn0] = fmaxf(a0[v] + b0, 0.f);
        E[(size_t)gm * OUT_DIM + gn1] = fmaxf(a1[v] + b1, 0.f);
      }
    }
  }
}

// ---------------- row squared norms ----------------------------------------
__device__ inline float block_reduce_sum(float v, float* sh) {
  int tid = threadIdx.x;
  sh[tid] = v;
  __syncthreads();
  for (int s = 128; s > 0; s >>= 1) {
    if (tid < s) sh[tid] += sh[tid + s];
    __syncthreads();
  }
  float r = sh[0];
  __syncthreads();
  return r;
}

__global__ void __launch_bounds__(256)
sqnorm_kernel(const float* __restrict__ E, float* __restrict__ norms) {
  __shared__ float sh[256];
  int row = blockIdx.x;
  float s = 0.f;
  for (int k = threadIdx.x; k < OUT_DIM; k += 256) {
    float v = E[(size_t)row * OUT_DIM + k];
    s += v * v;
  }
  s = block_reduce_sum(s, sh);
  if (threadIdx.x == 0) norms[row] = s;
}

// ---------------- GEMM 2/3: Euclidean distance matrices --------------------
__global__ void __launch_bounds__(256)
dist_gemm(const float* __restrict__ A, const float* __restrict__ B,
          const float* __restrict__ normA, const float* __restrict__ normB,
          float* __restrict__ D, int M, int N) {
  GEMM_SHARED_DECL
  int m0 = blockIdx.x * BM;
  int n0 = blockIdx.y * BN;
  int wave = threadIdx.x >> 5;
  int wm = wave >> 1, wn = wave & 1;
  v8f acc00 = {0, 0, 0, 0, 0, 0, 0, 0};
  v8f acc01 = acc00, acc10 = acc00, acc11 = acc00;
  for (int k0 = 0; k0 < OUT_DIM; k0 += BK) {
    load_tile_plain<BM>(A, OUT_DIM, m0, k0, M, sAh, sAl);
    load_tile_plain<BN>(B, OUT_DIM, n0, k0, N, sBh, sBl);
    __syncthreads();
    GEMM_WMMA_STEP(wm, wn, 0)
    GEMM_WMMA_STEP(wm, wn, 32)
    __syncthreads();
  }
  int lane = threadIdx.x & 31;
  int half = lane >> 4, nn = lane & 15;
  int gn0 = n0 + wn * 32 + nn;
  int gn1 = gn0 + 16;
  float nb0 = (gn0 < N) ? normB[gn0] : 0.f;
  float nb1 = (gn1 < N) ? normB[gn1] : 0.f;
#pragma unroll
  for (int ms = 0; ms < 2; ++ms) {
    int gm_base = m0 + wm * 32 + ms * 16 + half * 8;
    v8f a0 = ms ? acc10 : acc00;
    v8f a1 = ms ? acc11 : acc01;
#pragma unroll
    for (int v = 0; v < 8; ++v) {
      int gm = gm_base + v;
      if (gm < M) {
        float na = normA[gm];
        if (gn0 < N)
          D[(size_t)gm * N + gn0] = sqrtf(fmaxf(na + nb0 - 2.f * a0[v], 0.f));
        if (gn1 < N)
          D[(size_t)gm * N + gn1] = sqrtf(fmaxf(na + nb1 - 2.f * a1[v], 0.f));
      }
    }
  }
}

// ---------------- per-(class,row) reductions over 225 support cols ---------
__global__ void __launch_bounds__(256)
class_reduce(const float* __restrict__ Dqs, float* __restrict__ ave,
             int* __restrict__ pos, float* __restrict__ rowmax) {
  int c = blockIdx.x / NQT;
  int i = blockIdx.x % NQT;
  __shared__ float v[ST];
  __shared__ float amax[SHOT];
  int tid = threadIdx.x;
  if (tid < ST) v[tid] = Dqs[(size_t)i * NST + c * ST + tid];
  __syncthreads();
  if (tid < SHOT) {  // reshape(225)->(45,5), max over the 45 axis
    float m = -3.4e38f;
    for (int t2 = 0; t2 < T_TUP; ++t2) m = fmaxf(m, v[t2 * SHOT + tid]);
    amax[tid] = m;
  }
  __syncthreads();
  if (tid == 0) {
    float s = 0.f;
    for (int a = 0; a < SHOT; ++a) s += amax[a];
    ave[c * NQT + i] = s / (float)SHOT;
    float m = v[0];
    int mi = 0;
    for (int j = 1; j < ST; ++j)
      if (v[j] > m) { m = v[j]; mi = j; }   // first-occurrence argmax
    rowmax[c * NQT + i] = m;
    pos[c * NQT + i] = mi;
  }
}

// record[c][mp][s2] = #{ i : Dss[c*ST+pos[c,i]][oc*ST+s2] > ave[c,i] }
__global__ void __launch_bounds__(256)
record_kernel(const float* __restrict__ Dss, const int* __restrict__ pos,
              const float* __restrict__ ave, float* __restrict__ record) {
  int s2 = blockIdx.x, mp = blockIdx.y, c = blockIdx.z;
  int oc = mp + (mp >= c ? 1 : 0);
  int col = oc * ST + s2;
  __shared__ float sh[256];
  float cnt = 0.f;
  for (int i = threadIdx.x; i < NQT; i += 256) {
    int p = pos[c * NQT + i];
    float d = Dss[(size_t)(c * ST + p) * NST + col];
    if (d > ave[c * NQT + i]) cnt += 1.f;
  }
  cnt = block_reduce_sum(cnt, sh);
  if (threadIdx.x == 0) record[(c * (WAY - 1) + mp) * ST + s2] = cnt;
}

// threshold + mask + global mask-sum per class
__global__ void __launch_bounds__(256)
maskgen(const float* __restrict__ record, float* __restrict__ mask,
        float* __restrict__ msum) {
  int c = blockIdx.x, tid = threadIdx.x;
  __shared__ float sh[256];
  float total = 0.f;
  for (int mp = 0; mp < WAY - 1; ++mp) {
    float r = (tid < ST) ? record[(c * (WAY - 1) + mp) * ST + tid] : 0.f;
    float s = block_reduce_sum(r, sh);
    float nz = block_reduce_sum((tid < ST && r != 0.f) ? 1.f : 0.f, sh);
    float thr = s / fmaxf(nz, 1.f);
    float m = (tid < ST) ? ((r < thr) ? 1.f : 0.f) : 0.f;
    if (tid < ST) mask[(c * (WAY - 1) + mp) * ST + tid] = m;
    total += block_reduce_sum(m, sh);
  }
  if (tid == 0) msum[c] = fmaxf(total, 1.f);
}

// contrast[q][c] = mean_t [ sum_{mp,s2} Dqs * mask ] / msum[c] / (WAY-1)
__global__ void __launch_bounds__(256)
contrast_kernel(const float* __restrict__ Dqs, const float* __restrict__ mask,
                const float* __restrict__ msum, float* __restrict__ contrast) {
  int q = blockIdx.x, c = blockIdx.y, tid = threadIdx.x;
  __shared__ float mk[(WAY - 1) * ST];
  __shared__ int cb[WAY - 1];
  __shared__ float sh[256];
  for (int j = tid; j < (WAY - 1) * ST; j += 256)
    mk[j] = mask[c * (WAY - 1) * ST + j];
  if (tid < WAY - 1) cb[tid] = (tid + (tid >= c ? 1 : 0)) * ST;
  __syncthreads();
  float s = 0.f;
  const int NC = (WAY - 1) * ST;  // 900
  for (int idx = tid; idx < T_TUP * NC; idx += 256) {
    int t = idx / NC, j = idx - t * NC;
    float m = mk[j];
    if (m != 0.f) {
      int mp = j / ST, s2 = j - mp * ST;
      s += Dqs[(size_t)(q * T_TUP + t) * NST + cb[mp] + s2];
    }
  }
  s = block_reduce_sum(s, sh);
  if (tid == 0)
    contrast[q * WAY + c] = s / msum[c] / (float)T_TUP / (float)(WAY - 1);
}

// dist_max + logits_contrast, written to d_out
__global__ void __launch_bounds__(256)
finalize(const float* __restrict__ rowmax, const float* __restrict__ contrast,
         float* __restrict__ out) {
  int idx = blockIdx.x * 256 + threadIdx.x;
  if (idx >= NQ * WAY) return;
  int q = idx / WAY, c = idx - q * WAY;
  float s = 0.f;
  for (int t = 0; t < T_TUP; ++t) s += rowmax[c * NQT + q * T_TUP + t];
  float dm = s / (float)T_TUP;
  out[idx] = dm;
  out[NQ * WAY + idx] = dm / (contrast[idx] + dm);
}

// ---------------------------------------------------------------------------
extern "C" void kernel_launch(void* const* d_in, const int* in_sizes, int n_in,
                              void* d_out, int out_size, void* d_ws, size_t ws_size,
                              hipStream_t stream) {
  (void)in_sizes; (void)n_in; (void)out_size; (void)ws_size;
  const float* support = (const float*)d_in[0];   // [25,10,2048]
  const float* queries = (const float*)d_in[2];   // [75,10,2048]
  const float* W       = (const float*)d_in[3];   // [2048,4096]
  const float* bias    = (const float*)d_in[4];   // [2048]
  float* out = (float*)d_out;                     // 375 dist_max + 375 logits

  char* base = (char*)d_ws;
  size_t off = 0;
  auto wsAlloc = [&](size_t bytes) -> void* {
    void* p = base + off;
    off += (bytes + 511) & ~(size_t)511;
    return p;
  };
  float* E        = (float*)wsAlloc((size_t)NROWS * OUT_DIM * sizeof(float));
  float* norms    = (float*)wsAlloc((size_t)NROWS * sizeof(float));
  float* Dqs      = (float*)wsAlloc((size_t)NQT * NST * sizeof(float));
  float* Dss      = (float*)wsAlloc((size_t)NST * NST * sizeof(float));
  float* ave      = (float*)wsAlloc((size_t)WAY * NQT * sizeof(float));
  int*   pos      = (int*)wsAlloc((size_t)WAY * NQT * sizeof(int));
  float* rowmax   = (float*)wsAlloc((size_t)WAY * NQT * sizeof(float));
  float* record   = (float*)wsAlloc((size_t)WAY * (WAY - 1) * ST * sizeof(float));
  float* mask     = (float*)wsAlloc((size_t)WAY * (WAY - 1) * ST * sizeof(float));
  float* msum     = (float*)wsAlloc(WAY * sizeof(float));
  float* contrast = (float*)wsAlloc((size_t)NQ * WAY * sizeof(float));

  const float* Es = E + (size_t)NQT * OUT_DIM;
  const float* ns = norms + NQT;

  embed_gemm<<<dim3((NROWS + BM - 1) / BM, OUT_DIM / BN), 256, 0, stream>>>(
      queries, support, W, bias, E);
  sqnorm_kernel<<<NROWS, 256, 0, stream>>>(E, norms);
  dist_gemm<<<dim3((NQT + BM - 1) / BM, (NST + BN - 1) / BN), 256, 0, stream>>>(
      E, Es, norms, ns, Dqs, NQT, NST);
  dist_gemm<<<dim3((NST + BM - 1) / BM, (NST + BN - 1) / BN), 256, 0, stream>>>(
      Es, Es, ns, ns, Dss, NST, NST);
  class_reduce<<<WAY * NQT, 256, 0, stream>>>(Dqs, ave, pos, rowmax);
  record_kernel<<<dim3(ST, WAY - 1, WAY), 256, 0, stream>>>(Dss, pos, ave, record);
  maskgen<<<WAY, 256, 0, stream>>>(record, mask, msum);
  contrast_kernel<<<dim3(NQ, WAY), 256, 0, stream>>>(Dqs, mask, msum, contrast);
  finalize<<<(NQ * WAY + 255) / 256, 256, 0, stream>>>(rowmax, contrast, out);
}
